// GRBFLA_80212809220577
// MI455X (gfx1250) — matrix operand order repaired and checked by reading, not compiled, not run on hardware
//
#include <hip/hip_runtime.h>

// ---------------------------------------------------------------------------
// MI455X (gfx1250) fused block:
//   BN1 -> 1x1 conv -> DW3x3 -> BN2 -> 1x1 conv -> +x -> linear attention
//   -> proj -> +  -> LN -> MLP(GELU) -> +
// All large GEMMs via v_wmma_f32_16x16x32_bf16 (wave32, 16x16 tiles).
// GEMM: 128x128 block tile, 8 waves, 32x64 wave tile -> 8 WMMA / 6 frag loads
// per K-chunk (3:1 ds:wmma). global_prefetch_b8 hides single-buffer latency.
// ---------------------------------------------------------------------------

typedef __attribute__((ext_vector_type(16))) __bf16 v16bf;
typedef __attribute__((ext_vector_type(8)))  float  v8f;

#define NB   8
#define CCH  384
#define HH   56
#define WW   56
#define HWP  3136          // 56*56
#define PPIX 25088         // NB*HWP
#define NHEAD 8
#define HD    48

__device__ __forceinline__ unsigned short f2bf(float f) {
  unsigned u = __float_as_uint(f);
  unsigned r = u + 0x7FFFu + ((u >> 16) & 1u);
  return (unsigned short)(r >> 16);
}
__device__ __forceinline__ float bf2f(unsigned short h) {
  return __uint_as_float(((unsigned)h) << 16);
}

union FragB16 { unsigned u[8]; v16bf v; };

// A fragment: 16(M) x 32(K) bf16 from LDS row-major tile [m][k], stride even.
// ISA 7.12.2: lanes 0-15 m=lane, VGPR v<4 -> K=2v,2v+1 ; v>=4 -> 16+2(v-4),..;
// lanes 16-31: +8 on the K pair base.
__device__ __forceinline__ v16bf load_a_frag(const unsigned short* sm, int stride,
                                             int mBase, int lane) {
  int m = mBase + (lane & 15);
  int half = lane >> 4;
  const unsigned short* row = sm + m * stride;
  FragB16 f;
#pragma unroll
  for (int v = 0; v < 8; ++v) {
    int kb = (v < 4) ? (2 * v + 8 * half) : (16 + 2 * (v - 4) + 8 * half);
    f.u[v] = *(const unsigned*)(row + kb);
  }
  return f.v;
}

// B fragment: 32(K) x 16(N) bf16 from LDS tile stored TRANSPOSED as [n][k].
// lanes 0-15: n=lane, v<4 -> K=2v,2v+1 ; v>=4 -> 16+2(v-4),.. ; lanes 16-31: +8.
__device__ __forceinline__ v16bf load_b_frag(const unsigned short* smT, int stride,
                                             int nBase, int lane) {
  int n = nBase + (lane & 15);
  int half = lane >> 4;
  const unsigned short* row = smT + n * stride;
  FragB16 f;
#pragma unroll
  for (int v = 0; v < 8; ++v) {
    int kb = (v < 4) ? (2 * v + 8 * half) : (16 + 2 * (v - 4) + 8 * half);
    f.u[v] = *(const unsigned*)(row + kb);
  }
  return f.v;
}

__device__ __forceinline__ v8f wmma_bf16(v16bf a, v16bf b, v8f c) {
  return __builtin_amdgcn_wmma_f32_16x16x32_bf16(false, a, false, b,
                                                 (short)0, c, false, false);
}

// -------------------------- generic GEMM + epilogue -------------------------
#define FB_BIAS     1
#define FB_PHI      2
#define FB_GELU     4
#define FB_RES_ROW  8
#define FB_RES_NCHW 16
#define FB_OUT_BF16 32
#define FB_OUT_F32  64
#define FB_OUT_NCHW 128

template <int EPI>
__global__ __launch_bounds__(256) void gemm_bf16_k(
    const unsigned short* __restrict__ A, int lda,
    const unsigned short* __restrict__ Bm, int ldb,
    int M, int N, int K,
    const float* __restrict__ bias,
    const float* __restrict__ res, int ldr,
    unsigned short* __restrict__ outb, int ldob,
    float* __restrict__ outf, int ldof,
    int phiLimit) {
  constexpr int BM = 128, BN = 128, BK = 32, SK = BK + 2;
  __shared__ unsigned short sA[BM][SK];   // [m][k]
  __shared__ unsigned short sBT[BN][SK];  // [n][k]  (transposed B tile)

  const int tid = threadIdx.x;
  const int lane = tid & 31;
  const int wave = tid >> 5;       // 0..7
  const int waveM = wave >> 1;     // 4 wave-rows of 32
  const int waveN = wave & 1;      // 2 wave-cols of 64
  const int m0 = blockIdx.y * BM;
  const int n0 = blockIdx.x * BN;

  v8f acc[2][4] = {};

  for (int kk = 0; kk < K; kk += BK) {
    {  // A tile: 128 rows x 16 dwords; 8 dwords per thread
      int c2 = (tid & 15) * 2;
      int r0 = tid >> 4;
#pragma unroll
      for (int i = 0; i < 8; ++i) {
        int row = r0 + i * 16;
        *(unsigned*)&sA[row][c2] =
            *(const unsigned*)(A + (size_t)(m0 + row) * lda + kk + c2);
        if (kk + BK < K)  // prefetch next A tile (global_prefetch_b8)
          __builtin_prefetch(A + (size_t)(m0 + row) * lda + kk + BK + c2, 0, 0);
      }
    }
    {  // B tile: read row-major [k][n], write transposed [n][k]; 16 elem/thread
      int kr = tid >> 3;            // 0..31
      int cb = (tid & 7) * 16;      // 16 columns per thread
      const unsigned short* Brow = Bm + (size_t)(kk + kr) * ldb + n0 + cb;
#pragma unroll
      for (int j = 0; j < 16; ++j) sBT[cb + j][kr] = Brow[j];
      if (kk + BK < K)
        __builtin_prefetch(Bm + (size_t)(kk + BK + kr) * ldb + n0 + cb, 0, 0);
    }
    __syncthreads();

    v16bf af0 = load_a_frag(&sA[0][0], SK, waveM * 32, lane);
    v16bf af1 = load_a_frag(&sA[0][0], SK, waveM * 32 + 16, lane);
    v16bf bf0 = load_b_frag(&sBT[0][0], SK, waveN * 64, lane);
    v16bf bf1 = load_b_frag(&sBT[0][0], SK, waveN * 64 + 16, lane);
    v16bf bf2 = load_b_frag(&sBT[0][0], SK, waveN * 64 + 32, lane);
    v16bf bf3 = load_b_frag(&sBT[0][0], SK, waveN * 64 + 48, lane);
    acc[0][0] = wmma_bf16(af0, bf0, acc[0][0]);
    acc[0][1] = wmma_bf16(af0, bf1, acc[0][1]);
    acc[0][2] = wmma_bf16(af0, bf2, acc[0][2]);
    acc[0][3] = wmma_bf16(af0, bf3, acc[0][3]);
    acc[1][0] = wmma_bf16(af1, bf0, acc[1][0]);
    acc[1][1] = wmma_bf16(af1, bf1, acc[1][1]);
    acc[1][2] = wmma_bf16(af1, bf2, acc[1][2]);
    acc[1][3] = wmma_bf16(af1, bf3, acc[1][3]);
    __syncthreads();
  }

#pragma unroll
  for (int mi = 0; mi < 2; ++mi) {
#pragma unroll
    for (int ni = 0; ni < 4; ++ni) {
      int col = n0 + waveN * 64 + ni * 16 + (lane & 15);
#pragma unroll
      for (int r = 0; r < 8; ++r) {
        int row = m0 + waveM * 32 + mi * 16 + r + 8 * (lane >> 4);
        float val = acc[mi][ni][r];
        if (EPI & FB_BIAS) val += bias[col];
        if (EPI & FB_PHI) {
          if (col < phiLimit) val = expf(0.5f * (2.0f * val - val * val));
        }
        if (EPI & FB_GELU) val = 0.5f * val * (1.0f + erff(val * 0.70710678118f));
        if (EPI & FB_RES_ROW) val += res[(size_t)row * ldr + col];
        if (EPI & FB_RES_NCHW) {
          int b = row / HWP, s = row - b * HWP;
          val += res[((size_t)(b * CCH + col)) * HWP + s];
        }
        if (EPI & FB_OUT_F32) outf[(size_t)row * ldof + col] = val;
        if (EPI & FB_OUT_BF16) outb[(size_t)row * ldob + col] = f2bf(val);
        if (EPI & FB_OUT_NCHW) {
          int b = row / HWP, s = row - b * HWP;
          outf[((size_t)(b * CCH + col)) * HWP + s] = val;
        }
      }
    }
  }
}

// ------------------------- elementwise / prep kernels -----------------------
__global__ __launch_bounds__(256) void bn1_k(
    const float* __restrict__ x, const float* __restrict__ w,
    const float* __restrict__ bb, const float* __restrict__ m,
    const float* __restrict__ v, unsigned short* __restrict__ xb) {
  size_t gid = (size_t)blockIdx.x * 256 + threadIdx.x;  // NCHW order
  if (gid >= (size_t)NB * CCH * HWP) return;
  int s = (int)(gid % HWP);
  int c = (int)((gid / HWP) % CCH);
  int b = (int)(gid / ((size_t)HWP * CCH));
  float sc = w[c] * rsqrtf(v[c] + 1e-5f);
  float val = (x[gid] - m[c]) * sc + bb[c];
  xb[((size_t)b * HWP + s) * CCH + c] = f2bf(val);
}

__global__ __launch_bounds__(256) void dw_k(
    const unsigned short* __restrict__ h1, const float* __restrict__ dw,
    const float* __restrict__ db, const float* __restrict__ w2,
    const float* __restrict__ b2, const float* __restrict__ m2,
    const float* __restrict__ v2, unsigned short* __restrict__ d2) {
  size_t gid = (size_t)blockIdx.x * 256 + threadIdx.x;  // pix*C + c
  if (gid >= (size_t)PPIX * CCH) return;
  int c = (int)(gid % CCH);
  size_t pix = gid / CCH;
  int s = (int)(pix % HWP);
  int b = (int)(pix / HWP);
  int y = s / WW, xx = s - y * WW;
  float acc = 0.0f;
#pragma unroll
  for (int ky = 0; ky < 3; ++ky) {
    int yy = y + ky - 1;
    if (yy < 0 || yy >= HH) continue;
#pragma unroll
    for (int kx = 0; kx < 3; ++kx) {
      int xc = xx + kx - 1;
      if (xc < 0 || xc >= WW) continue;
      size_t p2 = ((size_t)b * HWP + yy * WW + xc) * CCH + c;
      acc += bf2f(h1[p2]) * dw[c * 9 + ky * 3 + kx];
    }
  }
  acc += db[c];
  float sc = w2[c] * rsqrtf(v2[c] + 1e-5f);
  acc = (acc - m2[c]) * sc + b2[c];
  d2[gid] = f2bf(acc);
}

__global__ __launch_bounds__(256) void wtr_k(const float* __restrict__ in,
                                             unsigned short* __restrict__ out,
                                             int O, int K) {
  int gid = blockIdx.x * 256 + threadIdx.x;
  if (gid >= O * K) return;
  int o = gid / K, k = gid - o * K;           // in is (O,K) row-major
  out[(size_t)k * O + o] = f2bf(in[gid]);     // out is (K,O)
}

__global__ __launch_bounds__(256) void wcv_k(const float* __restrict__ in,
                                             unsigned short* __restrict__ out,
                                             int n) {
  int gid = blockIdx.x * 256 + threadIdx.x;
  if (gid < n) out[gid] = f2bf(in[gid]);
}

// ------------------------- attention: kv = kp^T @ v -------------------------
__device__ __forceinline__ void store_kv_tile(v8f acc, int t, int bh, int lane,
                                              unsigned short* kvT) {
  int d0 = (t / 3) * 16, e0 = (t % 3) * 16;
  int e = e0 + (lane & 15);
#pragma unroll
  for (int r = 0; r < 8; ++r) {
    int d = d0 + r + 8 * (lane >> 4);
    kvT[((size_t)bh * HD + e) * 64 + d] = f2bf(acc[r]);  // kvT[e][d] = kv[d][e]
  }
}

__global__ __launch_bounds__(128) void attn_kv_k(
    const unsigned short* __restrict__ qkv,  // [P][1152]
    unsigned short* __restrict__ kvT,        // [64 bh][48 e][64 d] (d>=48 zero)
    float* __restrict__ ksum) {              // [64 bh][48]
  const int bh = blockIdx.x;
  const int b = bh >> 3, h = bh & 7;
  const int tid = threadIdx.x, lane = tid & 31, wave = tid >> 5;
  const size_t rowBase = (size_t)b * HWP * 1152;
  const int kcol = CCH + h * HD;      // k channels
  const int vcol = 2 * CCH + h * HD;  // v channels

  __shared__ unsigned short skp[HD][34];  // [d][n]  (A = kp^T, K=n contiguous)
  __shared__ unsigned short sv[HD][34];   // [e][n]  (B transposed layout)

  v8f a0 = {}, a1 = {}, a2 = {};
  const int t0 = wave, t1 = wave + 4, t2 = wave + 8;  // 9 tiles over 4 waves
  float myks = 0.0f;

  for (int nn = 0; nn < HWP; nn += 32) {
    for (int i = tid; i < 32 * HD; i += 128) {
      int n = i / HD, c = i - n * HD;
      const unsigned short* src = qkv + rowBase + (size_t)(nn + n) * 1152;
      skp[c][n] = src[kcol + c];
      sv[c][n] = src[vcol + c];
    }
    __syncthreads();
    if (tid < HD) {
      float s = 0.0f;
#pragma unroll
      for (int n = 0; n < 32; ++n) s += bf2f(skp[tid][n]);
      myks += s;
    }
    {
      v16bf af = load_a_frag(&skp[0][0], 34, (t0 / 3) * 16, lane);
      v16bf bf = load_b_frag(&sv[0][0], 34, (t0 % 3) * 16, lane);
      a0 = wmma_bf16(af, bf, a0);
    }
    {
      v16bf af = load_a_frag(&skp[0][0], 34, (t1 / 3) * 16, lane);
      v16bf bf = load_b_frag(&sv[0][0], 34, (t1 % 3) * 16, lane);
      a1 = wmma_bf16(af, bf, a1);
    }
    if (t2 < 9) {  // wave-uniform branch: EXEC stays all-ones for WMMA
      v16bf af = load_a_frag(&skp[0][0], 34, (t2 / 3) * 16, lane);
      v16bf bf = load_b_frag(&sv[0][0], 34, (t2 % 3) * 16, lane);
      a2 = wmma_bf16(af, bf, a2);
    }
    __syncthreads();
  }

  store_kv_tile(a0, t0, bh, lane, kvT);
  store_kv_tile(a1, t1, bh, lane, kvT);
  if (t2 < 9) store_kv_tile(a2, t2, bh, lane, kvT);
  if (tid < HD) ksum[bh * HD + tid] = myks;
  for (int i = tid; i < HD * 16; i += 128) {  // zero-pad d=48..63
    int e = i / 16, d = 48 + (i & 15);
    kvT[((size_t)bh * HD + e) * 64 + d] = 0;
  }
}

// z[n,h] = 1 / (qp . ksum + eps)
__global__ __launch_bounds__(256) void attn_z_k(
    const unsigned short* __restrict__ qkv, const float* __restrict__ ksum,
    float* __restrict__ z) {
  int gid = blockIdx.x * 256 + threadIdx.x;
  if (gid >= PPIX * NHEAD) return;
  int pix = gid >> 3, h = gid & 7;
  int b = pix / HWP;
  const unsigned short* q = qkv + (size_t)pix * 1152 + h * HD;  // phi(q) cols
  const float* ks = ksum + (b * NHEAD + h) * HD;
  float acc = 0.0f;
#pragma unroll
  for (int d = 0; d < HD; ++d) acc += bf2f(q[d]) * ks[d];
  z[gid] = 1.0f / (acc + 1e-6f);
}

// out[n, h*48+e] = (qp[n,:] . kv[:,e]) * z[n,h]
__global__ __launch_bounds__(128) void attn_out_k(
    const unsigned short* __restrict__ qkv,
    const unsigned short* __restrict__ kvT, const float* __restrict__ z,
    unsigned short* __restrict__ attn) {  // [P][384]
  const int m0 = blockIdx.x * 64;  // HWP % 64 == 0 -> tile within one batch
  const int h = blockIdx.y;
  const int b = m0 / HWP;
  const int bh = b * NHEAD + h;
  const int tid = threadIdx.x, lane = tid & 31, wave = tid >> 5;

  __shared__ unsigned short sq[64][66];   // [m][k], k padded 48->64 with zeros
  __shared__ unsigned short skv[HD][66];  // [e][d] = B transposed layout

  for (int i = tid; i < HD * 64; i += 128) {
    int e = i >> 6, d = i & 63;
    skv[e][d] = kvT[((size_t)bh * HD + e) * 64 + d];
  }
  for (int i = tid; i < 64 * 64; i += 128) {
    int m = i >> 6, k = i & 63;
    unsigned short vv = 0;
    if (k < HD) vv = qkv[(size_t)(m0 + m) * 1152 + h * HD + k];  // phi(q)
    sq[m][k] = vv;
  }
  __syncthreads();

  v8f acc0 = {}, acc1 = {}, acc2 = {};
#pragma unroll
  for (int kk = 0; kk < 64; kk += 32) {
    v16bf af = load_a_frag(&sq[0][kk], 66, wave * 16, lane);
    v16bf b0 = load_b_frag(&skv[0][kk], 66, 0, lane);
    v16bf b1 = load_b_frag(&skv[0][kk], 66, 16, lane);
    v16bf b2 = load_b_frag(&skv[0][kk], 66, 32, lane);
    acc0 = wmma_bf16(af, b0, acc0);
    acc1 = wmma_bf16(af, b1, acc1);
    acc2 = wmma_bf16(af, b2, acc2);
  }

#pragma unroll
  for (int sub = 0; sub < 3; ++sub) {
    v8f acc = (sub == 0) ? acc0 : (sub == 1) ? acc1 : acc2;
    int e = sub * 16 + (lane & 15);
#pragma unroll
    for (int r = 0; r < 8; ++r) {
      int row = m0 + wave * 16 + r + 8 * (lane >> 4);
      float val = acc[r] * z[row * NHEAD + h];
      attn[(size_t)row * CCH + h * HD + e] = f2bf(val);
    }
  }
}

// ------------------------------- LayerNorm ----------------------------------
__global__ __launch_bounds__(128) void ln_k(const float* __restrict__ x2,
                                            const float* __restrict__ lnw,
                                            const float* __restrict__ lnb,
                                            unsigned short* __restrict__ xn) {
  const int pix = blockIdx.x;
  const int tid = threadIdx.x;
  __shared__ float red[128];
  const float* row = x2 + (size_t)pix * CCH;
  float v0 = row[tid], v1 = row[tid + 128], v2 = row[tid + 256];
  red[tid] = v0 + v1 + v2;
  __syncthreads();
  for (int o = 64; o > 0; o >>= 1) {
    if (tid < o) red[tid] += red[tid + o];
    __syncthreads();
  }
  float mean = red[0] * (1.0f / CCH);
  __syncthreads();
  float d0 = v0 - mean, d1 = v1 - mean, d2 = v2 - mean;
  red[tid] = d0 * d0 + d1 * d1 + d2 * d2;
  __syncthreads();
  for (int o = 64; o > 0; o >>= 1) {
    if (tid < o) red[tid] += red[tid + o];
    __syncthreads();
  }
  float rstd = rsqrtf(red[0] * (1.0f / CCH) + 1e-5f);
  unsigned short* out = xn + (size_t)pix * CCH;
  out[tid] = f2bf(d0 * rstd * lnw[tid] + lnb[tid]);
  out[tid + 128] = f2bf(d1 * rstd * lnw[tid + 128] + lnb[tid + 128]);
  out[tid + 256] = f2bf(d2 * rstd * lnw[tid + 256] + lnb[tid + 256]);
}

// ------------------------------- launcher -----------------------------------
extern "C" void kernel_launch(void* const* d_in, const int* in_sizes, int n_in,
                              void* d_out, int out_size, void* d_ws,
                              size_t ws_size, hipStream_t stream) {
  const float* x      = (const float*)d_in[0];
  const float* bn1_w  = (const float*)d_in[1];
  const float* bn1_b  = (const float*)d_in[2];
  const float* bn1_m  = (const float*)d_in[3];
  const float* bn1_v  = (const float*)d_in[4];
  const float* cw1    = (const float*)d_in[5];
  const float* cb1    = (const float*)d_in[6];
  const float* dwv    = (const float*)d_in[7];
  const float* dbv    = (const float*)d_in[8];
  const float* bn2_w  = (const float*)d_in[9];
  const float* bn2_b  = (const float*)d_in[10];
  const float* bn2_m  = (const float*)d_in[11];
  const float* bn2_v  = (const float*)d_in[12];
  const float* cw2    = (const float*)d_in[13];
  const float* cb2    = (const float*)d_in[14];
  const float* qkv_w  = (const float*)d_in[15];
  const float* proj_w = (const float*)d_in[16];
  const float* proj_b = (const float*)d_in[17];
  const float* ln_w   = (const float*)d_in[18];
  const float* ln_b   = (const float*)d_in[19];
  const float* mlp_w1 = (const float*)d_in[20];
  const float* mlp_b1 = (const float*)d_in[21];
  const float* mlp_w2 = (const float*)d_in[22];
  const float* mlp_b2 = (const float*)d_in[23];
  (void)in_sizes; (void)n_in; (void)out_size; (void)ws_size;

  char* ws = (char*)d_ws;
  size_t off = 0;
  auto alloc = [&](size_t bytes) -> void* {
    void* p = ws + off;
    off += (bytes + 255) & ~(size_t)255;
    return p;
  };
  const size_t P = PPIX;
  unsigned short* cw1T  = (unsigned short*)alloc(384 * 384 * 2);
  unsigned short* cw2T  = (unsigned short*)alloc(384 * 384 * 2);
  unsigned short* qkvT  = (unsigned short*)alloc(384 * 1152 * 2);
  unsigned short* projT = (unsigned short*)alloc(384 * 384 * 2);
  unsigned short* w1b   = (unsigned short*)alloc(384 * 1536 * 2);
  unsigned short* w2b   = (unsigned short*)alloc(1536 * 384 * 2);
  unsigned short* bufA  = (unsigned short*)alloc(P * 1536 * 2);  // qkv, then hmid
  unsigned short* xb    = (unsigned short*)alloc(P * 384 * 2);   // bn1 out; later attn out
  unsigned short* h1    = (unsigned short*)alloc(P * 384 * 2);   // gemm1 out; later xn
  unsigned short* d2    = (unsigned short*)alloc(P * 384 * 2);   // depthwise+bn2 out
  unsigned short* x1b   = (unsigned short*)alloc(P * 384 * 2);   // x1 bf16
  float* x1f  = (float*)alloc(P * 384 * 4);
  float* x2f  = (float*)alloc(P * 384 * 4);
  unsigned short* kvT = (unsigned short*)alloc(64 * 48 * 64 * 2);
  float* ksum = (float*)alloc(64 * 48 * 4);
  float* zbuf = (float*)alloc(P * 8 * 4);
  unsigned short* attn = xb;  // alias: xb dead after GEMM1
  unsigned short* xn   = h1;  // alias: h1 dead after depthwise

  // weights -> bf16 (transposed to [K][N] where needed)
  wtr_k<<<(384 * 384 + 255) / 256, 256, 0, stream>>>(cw1, cw1T, 384, 384);
  wtr_k<<<(384 * 384 + 255) / 256, 256, 0, stream>>>(cw2, cw2T, 384, 384);
  wtr_k<<<(1152 * 384 + 255) / 256, 256, 0, stream>>>(qkv_w, qkvT, 1152, 384);
  wtr_k<<<(384 * 384 + 255) / 256, 256, 0, stream>>>(proj_w, projT, 384, 384);
  wcv_k<<<(384 * 1536 + 255) / 256, 256, 0, stream>>>(mlp_w1, w1b, 384 * 1536);
  wcv_k<<<(1536 * 384 + 255) / 256, 256, 0, stream>>>(mlp_w2, w2b, 1536 * 384);

  const int nElem = NB * CCH * HWP;
  bn1_k<<<(nElem + 255) / 256, 256, 0, stream>>>(x, bn1_w, bn1_b, bn1_m, bn1_v, xb);

  // 128x128 block tiles: P/128 = 196 row-blocks
  dim3 g3(384 / 128, P / 128), g9(1152 / 128, P / 128), g12(1536 / 128, P / 128);

  // h1 = bn1(x) @ cw1^T + cb1
  gemm_bf16_k<FB_BIAS | FB_OUT_BF16><<<g3, 256, 0, stream>>>(
      xb, 384, cw1T, 384, P, 384, 384, cb1, nullptr, 0, h1, 384, nullptr, 0, 0);

  // depthwise 3x3 + db + bn2
  dw_k<<<(nElem + 255) / 256, 256, 0, stream>>>(h1, dwv, dbv, bn2_w, bn2_b,
                                                bn2_m, bn2_v, d2);

  // x1 = d2 @ cw2^T + cb2 + x   (store f32 + bf16)
  gemm_bf16_k<FB_BIAS | FB_RES_NCHW | FB_OUT_F32 | FB_OUT_BF16>
      <<<g3, 256, 0, stream>>>(d2, 384, cw2T, 384, P, 384, 384, cb2, x, 0,
                               x1b, 384, x1f, 384, 0);

  // qkv = x1 @ qkv_w^T ; phi() applied to q,k columns (< 768)
  gemm_bf16_k<FB_PHI | FB_OUT_BF16><<<g9, 256, 0, stream>>>(
      x1b, 384, qkvT, 1152, P, 1152, 384, nullptr, nullptr, 0, bufA, 1152,
      nullptr, 0, 768);

  // kv[d,e] = sum_n phi(k)[n,d] v[n,e] ; ksum[d] = sum_n phi(k)[n,d]
  attn_kv_k<<<64, 128, 0, stream>>>(bufA, kvT, ksum);
  attn_z_k<<<(PPIX * 8 + 255) / 256, 256, 0, stream>>>(bufA, ksum, zbuf);
  attn_out_k<<<dim3(P / 64, NHEAD), 128, 0, stream>>>(bufA, kvT, zbuf, attn);

  // x2 = x1 + attn @ proj_w^T + proj_b  (f32)
  gemm_bf16_k<FB_BIAS | FB_RES_ROW | FB_OUT_F32><<<g3, 256, 0, stream>>>(
      attn, 384, projT, 384, P, 384, 384, proj_b, x1f, 384, nullptr, 0, x2f,
      384, 0);

  // LayerNorm over channels
  ln_k<<<P, 128, 0, stream>>>(x2f, ln_w, ln_b, xn);

  // hmid = gelu(xn @ mlp_w1 + b1)
  gemm_bf16_k<FB_BIAS | FB_GELU | FB_OUT_BF16><<<g12, 256, 0, stream>>>(
      xn, 384, w1b, 1536, P, 1536, 384, mlp_b1, nullptr, 0, bufA, 1536,
      nullptr, 0, 0);

  // out = x2 + hmid @ mlp_w2 + b2, stored f32 NCHW
  gemm_bf16_k<FB_BIAS | FB_RES_ROW | FB_OUT_NCHW><<<g3, 256, 0, stream>>>(
      bufA, 1536, w2b, 384, P, 384, 1536, mlp_b2, x2f, 384, nullptr, 0,
      (float*)d_out, 0, 0);
}
